// ISAC_33062658244828
// MI455X (gfx1250) — matrix operand order repaired and checked by reading, not compiled, not run on hardware
//
#include <hip/hip_runtime.h>
#include <hip/hip_bf16.h>

typedef __attribute__((ext_vector_type(16))) _Float16 v16h;
typedef __attribute__((ext_vector_type(8)))  float    v8f;
typedef __attribute__((ext_vector_type(4)))  int      v4i;

#define NB      32      // batches
#define NC      40      // channels
#define NK      128     // taps (= stride)
#define NL      16000   // signal length (= FFT length)
#define NM      125     // decimated outputs per (b,c)
#define MTILES  8       // ceil(125/16)
#define CTILES  3       // ceil(40/16)
#define NTILE   (CTILES * 4)        // 12 (ctile, kstep) kernel tiles
#define WS_IMAG (NTILE * 32 * 16)   // halves: imag region offset (6144)
#define WIN     2176    // staged x window: 128 lookback + 16*128 span

#define GAS __attribute__((address_space(1)))
#define LAS __attribute__((address_space(3)))

// ---------------------------------------------------------------------------
// Kernel 1: pre-pack complex kernels to f16 in the exact WMMA A-operand lane
// layout (16-bit A 16x32: per lane two contiguous 8-tap runs kbA+0..7 and
// 16+kbA+0..7, kbA = 8*(lane>=16), row = lane&15). One 32B vector per
// (tile, lane, {r|i}) -> main kernel loads A with 2x b128, no conversions.
// ---------------------------------------------------------------------------
__global__ __launch_bounds__(32)
void ISAC_prepack(const float* __restrict__ kr, const float* __restrict__ ki,
                  _Float16* __restrict__ ws)
{
    const int lane = threadIdx.x & 31;
    const int half = lane >> 4;
    const int l15  = lane & 15;
    const int tile = blockIdx.x;          // ct*4 + ks, 0..11
    const int ct   = tile >> 2;
    const int ks   = tile & 3;
    const int kk   = ks * 32;
    const int kbA  = half * 8;

    const int c  = ct * 16 + l15;
    const int cc = (c < NC) ? c : (NC - 1);     // clamp: rows c>=40 never stored
    const float* __restrict__ rowr = kr + (size_t)cc * NK + kk + kbA;
    const float* __restrict__ rowi = ki + (size_t)cc * NK + kk + kbA;

    float4 r0 = *reinterpret_cast<const float4*>(rowr + 0);
    float4 r1 = *reinterpret_cast<const float4*>(rowr + 4);
    float4 r2 = *reinterpret_cast<const float4*>(rowr + 16);
    float4 r3 = *reinterpret_cast<const float4*>(rowr + 20);
    float4 i0 = *reinterpret_cast<const float4*>(rowi + 0);
    float4 i1 = *reinterpret_cast<const float4*>(rowi + 4);
    float4 i2 = *reinterpret_cast<const float4*>(rowi + 16);
    float4 i3 = *reinterpret_cast<const float4*>(rowi + 20);

    v16h ar, ai;
    ar[0]=(_Float16)r0.x; ar[1]=(_Float16)r0.y; ar[2]=(_Float16)r0.z; ar[3]=(_Float16)r0.w;
    ar[4]=(_Float16)r1.x; ar[5]=(_Float16)r1.y; ar[6]=(_Float16)r1.z; ar[7]=(_Float16)r1.w;
    ar[8]=(_Float16)r2.x; ar[9]=(_Float16)r2.y; ar[10]=(_Float16)r2.z; ar[11]=(_Float16)r2.w;
    ar[12]=(_Float16)r3.x; ar[13]=(_Float16)r3.y; ar[14]=(_Float16)r3.z; ar[15]=(_Float16)r3.w;
    ai[0]=(_Float16)i0.x; ai[1]=(_Float16)i0.y; ai[2]=(_Float16)i0.z; ai[3]=(_Float16)i0.w;
    ai[4]=(_Float16)i1.x; ai[5]=(_Float16)i1.y; ai[6]=(_Float16)i1.z; ai[7]=(_Float16)i1.w;
    ai[8]=(_Float16)i2.x; ai[9]=(_Float16)i2.y; ai[10]=(_Float16)i2.z; ai[11]=(_Float16)i2.w;
    ai[12]=(_Float16)i3.x; ai[13]=(_Float16)i3.y; ai[14]=(_Float16)i3.z; ai[15]=(_Float16)i3.w;

    const size_t off = ((size_t)tile * 32 + lane) * 16;   // halves
    *reinterpret_cast<v16h*>(ws + off)           = ar;    // 32B-aligned
    *reinterpret_cast<v16h*>(ws + WS_IMAG + off) = ai;
}

// ---------------------------------------------------------------------------
// Kernel 2: one wave per (batch b, 16-wide m-tile). Async-copy the circular
// x window into LDS (ASYNCcnt path), then Y[c,m] += Kr/Ki(16x32) x Xg(32x16)
// on v_wmma_f32_16x16x32_f16: 4 K-steps, 3 C-tiles x {real,imag} sharing one
// B operand. A operands come pre-packed f16 from ws (2x b128 each, no cvt).
// ---------------------------------------------------------------------------
__global__ __launch_bounds__(32)
void ISAC_fftconv_wmma(const float* __restrict__ x,
                       const _Float16* __restrict__ ws,
                       float* __restrict__ out)
{
    __shared__ float lx[WIN];

    const int lane = threadIdx.x & 31;
    const int half = lane >> 4;
    const int l15  = lane & 15;

    const int b  = blockIdx.x >> 3;
    const int m0 = (blockIdx.x & 7) << 4;

    const float* __restrict__ xb = x + (size_t)b * NL;

    // ---- stage x window: lx[j] = x[(m0*128 - 128 + j) mod 16000], j=0..2175
    // 544 float4 chunks, wrap at float4 granularity, window start 512B-aligned.
    const int wbase = m0 * NK - NK;
    #pragma unroll
    for (int it = 0; it < 17; ++it) {
        int j4 = (it * 32 + lane) * 4;
        int g  = wbase + j4;
        g = (g < 0)   ? g + NL : g;
        g = (g >= NL) ? g - NL : g;
#if __has_builtin(__builtin_amdgcn_global_load_async_to_lds_b128)
        __builtin_amdgcn_global_load_async_to_lds_b128(
            (GAS v4i*)(xb + g), (LAS v4i*)&lx[j4], 0, 0);
#else
        {   // inline-asm async copy: VDST = LDS byte offset, VADDR = 64b global
            unsigned int laddr =
                (unsigned int)(__UINTPTR_TYPE__)(LAS void*)&lx[j4];
            unsigned long long gaddr = (unsigned long long)(__UINTPTR_TYPE__)(xb + g);
            asm volatile("global_load_async_to_lds_b128 %0, %1, off"
                         :: "v"(laddr), "v"(gaddr) : "memory");
        }
#endif
    }
#if __has_builtin(__builtin_amdgcn_s_wait_asynccnt)
    __builtin_amdgcn_s_wait_asynccnt(0);
#else
    asm volatile("s_wait_asynccnt 0x0" ::: "memory");
#endif
    __syncthreads();

    // accumulators: [ctile][0=real,1=imag], 16x16 f32 each (8 VGPRs)
    v8f acc[CTILES][2] = {};

    const int kbB = half * 8;    // 16-bit B 32x16 layout: K = kbB + e (e<8), 16 + kbB + e-8 (e>=8)

    #pragma unroll
    for (int ks = 0; ks < 4; ++ks) {
        const int kk = ks * 32;

        // ---- B operand from LDS: Xg[k,m] = lx[128*(m-m0) + 128 - k]
        // jtop in [16, 2048]: always in-window, unconditional reads.
        const int jtop = 128 * l15 + 128 - kk - kbB;
        v16h bmat;
        #pragma unroll
        for (int e = 0; e < 16; ++e)
            bmat[e] = (_Float16)lx[jtop - e - ((e >= 8) ? 8 : 0)];

        // ---- pre-packed A operands; same B reused across 3 C-tiles x {r,i}
        #pragma unroll
        for (int ct = 0; ct < CTILES; ++ct) {
            const size_t off = ((size_t)(ct * 4 + ks) * 32 + lane) * 16;
            v16h amr = *reinterpret_cast<const v16h*>(ws + off);
            v16h ami = *reinterpret_cast<const v16h*>(ws + WS_IMAG + off);

            acc[ct][0] = __builtin_amdgcn_wmma_f32_16x16x32_f16(
                false, amr, false, bmat, (short)0, acc[ct][0], false, false);
            acc[ct][1] = __builtin_amdgcn_wmma_f32_16x16x32_f16(
                false, ami, false, bmat, (short)0, acc[ct][1], false, false);
        }
    }

    // ---- store interleaved complex64: out[((b*40 + c)*125 + m)] = {re, im}
    const int m = m0 + l15;
    if (m < NM) {
        #pragma unroll
        for (int ct = 0; ct < CTILES; ++ct) {
            #pragma unroll
            for (int v = 0; v < 8; ++v) {
                const int c = ct * 16 + v + 8 * half;   // f32 16x16 D-row map
                if (ct < 2 || half == 0) {              // ct==2 valid iff half==0
                    size_t idx = (((size_t)b * NC + c) * NM + m) * 2;
                    float2 val;
                    val.x = acc[ct][0][v];
                    val.y = acc[ct][1][v];
                    *reinterpret_cast<float2*>(out + idx) = val;
                }
            }
        }
    }
}

extern "C" void kernel_launch(void* const* d_in, const int* in_sizes, int n_in,
                              void* d_out, int out_size, void* d_ws, size_t ws_size,
                              hipStream_t stream) {
    const float* x  = (const float*)d_in[0];   // (32, 1, 16000) f32
    const float* kr = (const float*)d_in[1];   // (40, 128) f32
    const float* ki = (const float*)d_in[2];   // (40, 128) f32
    float* out = (float*)d_out;                // (32, 40, 125) complex64 interleaved
    _Float16* ws = (_Float16*)d_ws;            // 24576 B: pre-packed f16 A tiles

    ISAC_prepack<<<dim3(NTILE), dim3(32), 0, stream>>>(kr, ki, ws);
    ISAC_fftconv_wmma<<<dim3(NB * MTILES), dim3(32), 0, stream>>>(x, ws, out);
}
